// PartSegmentationEmbHead_18949395710667
// MI455X (gfx1250) — matrix operand order repaired and checked by reading, not compiled, not run on hardware
//
#include <hip/hip_runtime.h>
#include <stdint.h>

// Problem constants (match reference)
#define B_ 4
#define N_ 8192
#define G_ 512
#define E_ 512
#define C_ 512
#define MT_ 32                 // points per workgroup (2 m-subtiles of 16)

typedef __attribute__((ext_vector_type(16))) __bf16 v16bf;
typedef __attribute__((ext_vector_type(8)))  float  v8f;

static __device__ __forceinline__ uint16_t f2bf(float f) {
    uint32_t u = __float_as_uint(f);
    u += 0x7FFFu + ((u >> 16) & 1u);   // round-to-nearest-even
    return (uint16_t)(u >> 16);
}

// ---------------------------------------------------------------------------
// Fold BatchNorm (eval) + conv bias into per-channel alpha/beta:
//   y = x*alpha + beta,  alpha = g*rsqrt(v+eps),  beta = be + (b - m)*alpha
// bn layout: [alpha1(512) | beta1(512) | alpha2(512) | beta2(512)]
// ---------------------------------------------------------------------------
__global__ void k_bn(const float* __restrict__ b1, const float* __restrict__ g1,
                     const float* __restrict__ be1, const float* __restrict__ m1,
                     const float* __restrict__ v1,
                     const float* __restrict__ b2, const float* __restrict__ g2,
                     const float* __restrict__ be2, const float* __restrict__ m2,
                     const float* __restrict__ v2,
                     float* __restrict__ bn) {
    int c = blockIdx.x * 256 + threadIdx.x;
    if (c >= C_) return;
    float a1 = g1[c] * rsqrtf(v1[c] + 1e-5f);
    bn[c]        = a1;
    bn[C_ + c]   = be1[c] + (b1[c] - m1[c]) * a1;
    float a2 = g2[c] * rsqrtf(v2[c] + 1e-5f);
    bn[2*C_ + c] = a2;
    bn[3*C_ + c] = be2[c] + (b2[c] - m2[c]) * a2;
}

// ---------------------------------------------------------------------------
// Pre-swizzle weight matrix W[512, K] (row-major f32) into bf16 fragments in
// B-matrix (32 x 16) WMMA layout, fragment-major:
//   dst[(kb*32 + nb)*512 + lane*16 + e]
// ISA B layout: lanes 0-15 -> n=lane, K=0..7,16..23; lanes 16-31 -> n=lane-16,
// K=8..15,24..31. Each lane's fragment is one contiguous 32B chunk.
// ---------------------------------------------------------------------------
__global__ void k_swz(const float* __restrict__ w, uint16_t* __restrict__ ws,
                      int K, int total) {
    int t = blockIdx.x * 256 + threadIdx.x;
    if (t >= total) return;
    int e    = t & 15;
    int lane = (t >> 4) & 31;
    int nb   = (t >> 9) & 31;        // 512/16 = 32 n-tiles
    int kb   = t >> 14;
    int k32  = (e & 7) + ((lane >> 4) << 3) + ((e >> 3) << 4);
    int n    = nb * 16 + (lane & 15);
    ws[t] = f2bf(w[(size_t)n * K + kb * 32 + k32]);
}

// ---------------------------------------------------------------------------
// 3-NN + inverse-distance weights. One thread per point; centers in LDS.
// ---------------------------------------------------------------------------
__global__ __launch_bounds__(256) void k_knn(const float* __restrict__ xyz,
                                             const float* __restrict__ centers,
                                             int* __restrict__ idx,
                                             float* __restrict__ wt) {
    __shared__ float sc[G_ * 3];
    int b = blockIdx.x >> 5;                        // N/256 = 32 blocks/batch
    int p = ((blockIdx.x & 31) << 8) + threadIdx.x;
    for (int i = threadIdx.x; i < G_ * 3; i += 256)
        sc[i] = centers[(size_t)b * G_ * 3 + i];
    __syncthreads();
    const float* xp = xyz + ((size_t)b * N_ + p) * 3;
    float px = xp[0], py = xp[1], pz = xp[2];
    float d0 = 1e30f, d1 = 1e30f, d2 = 1e30f;
    int   i0 = 0, i1 = 0, i2 = 0;
    for (int g = 0; g < G_; ++g) {
        float dx = px - sc[3*g], dy = py - sc[3*g+1], dz = pz - sc[3*g+2];
        float d = dx*dx + dy*dy + dz*dz;
        if (d < d0)      { d2=d1; i2=i1; d1=d0; i1=i0; d0=d; i0=g; }
        else if (d < d1) { d2=d1; i2=i1; d1=d;  i1=g; }
        else if (d < d2) { d2=d;  i2=g; }
    }
    float r0 = 1.f/(d0+1e-8f), r1 = 1.f/(d1+1e-8f), r2 = 1.f/(d2+1e-8f);
    float rs = 1.f/(r0+r1+r2);
    size_t o = ((size_t)b * N_ + p) * 3;
    idx[o] = i0; idx[o+1] = i1; idx[o+2] = i2;
    wt[o] = r0*rs; wt[o+1] = r1*rs; wt[o+2] = r2*rs;
}

// ---------------------------------------------------------------------------
// Fused: gather-blend -> GEMM1 (1536->512, bf16 WMMA, f32 acc) -> BN1/ReLU
//        -> GEMM2 (512->512) -> BN2/ReLU -> out.
// 32 points per workgroup (2 m-subtiles); 8 waves x (2m x 4n) accumulators
// cover all 2x32 output tiles per layer. Each B-fragment global load feeds
// two WMMAs; each A-fragment LDS load feeds four.
// LDS: sA = one 32x256 activation chunk in A-fragment order (16 KB)
//      sH = full 32x512 bf16 h1 in A-fragment order for layer 2 (32 KB)
// A-fragment layout (16-bit, ISA 7.12.2): lane = m + 16*((k32>>3)&1),
//   halfword index e = ((k32&16)>>1) | (k32&7).
// ---------------------------------------------------------------------------
__global__ __launch_bounds__(256) void k_fused(
        const float* __restrict__ H4, const float* __restrict__ H8,
        const float* __restrict__ H12,
        const int* __restrict__ idx, const float* __restrict__ wt,
        const uint16_t* __restrict__ w1s, const uint16_t* __restrict__ w2s,
        const float* __restrict__ bn, float* __restrict__ out) {
    __shared__ __align__(32) uint16_t sA[16 * 512];   // (ms*8 + kb) fragments
    __shared__ __align__(32) uint16_t sH[32 * 512];   // (ms*16 + kb) fragments
    __shared__ float sW[MT_ * 3];
    __shared__ int   sI[MT_ * 3];

    const int b    = blockIdx.x / (N_ / MT_);
    const int p0   = (blockIdx.x % (N_ / MT_)) * MT_;
    const int tid  = threadIdx.x;
    const int wv   = tid >> 5;          // wave32: 8 waves
    const int lane = tid & 31;

    if (tid < MT_ * 3) {
        size_t o = ((size_t)b * N_ + p0) * 3 + tid;
        sI[tid] = idx[o];
        sW[tid] = wt[o];
    }
    __syncthreads();

    v8f acc[2][4];
    #pragma unroll
    for (int ms = 0; ms < 2; ++ms)
        #pragma unroll
        for (int j = 0; j < 4; ++j) acc[ms][j] = 0;

    // Staging assignment: 32 points x 256 K per chunk, 8192 elems / 256 thr.
    const int m   = tid >> 3;           // point row handled by this thread
    const int kl0 = (tid & 7) << 5;     // 32 consecutive K values per thread
    const int ms_ = m >> 4, mm_ = m & 15;

    for (int kc = 0; kc < 6; ++kc) {    // K chunks of 256 (2 per source tensor)
        const float* src = (kc < 2 ? H4 : (kc < 4 ? H8 : H12)) + (size_t)b * G_ * E_;
        const int koff = (kc & 1) ? 256 : 0;
        const float* r0 = src + (size_t)sI[m*3+0] * E_ + koff + kl0;
        const float* r1 = src + (size_t)sI[m*3+1] * E_ + koff + kl0;
        const float* r2 = src + (size_t)sI[m*3+2] * E_ + koff + kl0;
        const float w0 = sW[m*3+0], w1v = sW[m*3+1], w2v = sW[m*3+2];
        #pragma unroll
        for (int i = 0; i < 32; ++i) {
            float v = w0 * r0[i] + w1v * r1[i] + w2v * r2[i];
            int kl = kl0 + i, kb = kl >> 5, k32 = kl & 31;
            int ln = mm_ + (((k32 >> 3) & 1) << 4);
            int e  = ((k32 & 16) >> 1) | (k32 & 7);
            sA[(ms_ * 8 + kb) * 512 + ln * 16 + e] = f2bf(v);
        }
        __syncthreads();
        #pragma unroll
        for (int kb = 0; kb < 8; ++kb) {
            v16bf af0 = *(const v16bf*)&sA[ kb      * 512 + lane * 16];
            v16bf af1 = *(const v16bf*)&sA[(8 + kb) * 512 + lane * 16];
            const int kbg = kc * 8 + kb;
            #pragma unroll
            for (int j = 0; j < 4; ++j) {
                const int nb = wv + j * 8;
                v16bf bf = *(const v16bf*)&w1s[(size_t)(kbg * 32 + nb) * 512 + lane * 16];
                acc[0][j] = __builtin_amdgcn_wmma_f32_16x16x32_bf16(false, af0, false, bf, (short)0, acc[0][j], false, false);
                acc[1][j] = __builtin_amdgcn_wmma_f32_16x16x32_bf16(false, af1, false, bf, (short)0, acc[1][j], false, false);
            }
        }
        __syncthreads();
    }

    // BN1 + ReLU, restage h1 (bf16) into sH in layer-2 A-fragment order.
    // C layout: VGPR r holds (m = r + 8*(lane>=16), n = lane&15) of each tile.
    const int mb = (lane >> 4) << 3;
    const int nl = lane & 15;
    #pragma unroll
    for (int j = 0; j < 4; ++j) {
        const int nb  = wv + j * 8;
        const int col = nb * 16 + nl;          // layer-1 out channel == layer-2 K
        const float al = bn[col], bt = bn[512 + col];
        const int kb2 = col >> 5, k32 = col & 31;
        const int grp = ((k32 >> 3) & 1) << 4;
        const int e   = ((k32 & 16) >> 1) | (k32 & 7);
        #pragma unroll
        for (int ms = 0; ms < 2; ++ms) {
            #pragma unroll
            for (int r = 0; r < 8; ++r) {
                float v = acc[ms][j][r] * al + bt;
                v = v > 0.f ? v : 0.f;
                sH[(ms * 16 + kb2) * 512 + (mb + r + grp) * 16 + e] = f2bf(v);
            }
        }
    }
    __syncthreads();

    // Layer 2: K = 512 -> 16 bf16 fragments per m-subtile, LDS-resident A.
    v8f oacc[2][4];
    #pragma unroll
    for (int ms = 0; ms < 2; ++ms)
        #pragma unroll
        for (int j = 0; j < 4; ++j) oacc[ms][j] = 0;
    #pragma unroll
    for (int kb = 0; kb < 16; ++kb) {
        v16bf af0 = *(const v16bf*)&sH[ kb       * 512 + lane * 16];
        v16bf af1 = *(const v16bf*)&sH[(16 + kb) * 512 + lane * 16];
        #pragma unroll
        for (int j = 0; j < 4; ++j) {
            const int nb = wv + j * 8;
            v16bf bf = *(const v16bf*)&w2s[(size_t)(kb * 32 + nb) * 512 + lane * 16];
            oacc[0][j] = __builtin_amdgcn_wmma_f32_16x16x32_bf16(false, af0, false, bf, (short)0, oacc[0][j], false, false);
            oacc[1][j] = __builtin_amdgcn_wmma_f32_16x16x32_bf16(false, af1, false, bf, (short)0, oacc[1][j], false, false);
        }
    }

    // BN2 + ReLU + coalesced f32 store (lanes 0-15 write 16 consecutive cols).
    #pragma unroll
    for (int j = 0; j < 4; ++j) {
        const int nb  = wv + j * 8;
        const int col = nb * 16 + nl;
        const float al = bn[1024 + col], bt = bn[1536 + col];
        #pragma unroll
        for (int ms = 0; ms < 2; ++ms) {
            #pragma unroll
            for (int r = 0; r < 8; ++r) {
                float v = oacc[ms][j][r] * al + bt;
                v = v > 0.f ? v : 0.f;
                out[((size_t)b * N_ + p0 + ms * 16 + mb + r) * 512 + col] = v;
            }
        }
    }
}

// ---------------------------------------------------------------------------
// Host launcher
// ---------------------------------------------------------------------------
extern "C" void kernel_launch(void* const* d_in, const int* in_sizes, int n_in,
                              void* d_out, int out_size, void* d_ws, size_t ws_size,
                              hipStream_t stream) {
    const float* xyz     = (const float*)d_in[0];
    const float* centers = (const float*)d_in[1];
    const float* H4      = (const float*)d_in[2];
    const float* H8      = (const float*)d_in[3];
    const float* H12     = (const float*)d_in[4];
    const float* w1      = (const float*)d_in[5];
    const float* b1      = (const float*)d_in[6];
    const float* g1      = (const float*)d_in[7];
    const float* be1     = (const float*)d_in[8];
    const float* m1      = (const float*)d_in[9];
    const float* v1      = (const float*)d_in[10];
    const float* w2      = (const float*)d_in[11];
    const float* b2      = (const float*)d_in[12];
    const float* g2      = (const float*)d_in[13];
    const float* be2     = (const float*)d_in[14];
    const float* m2      = (const float*)d_in[15];
    const float* v2      = (const float*)d_in[16];

    // Workspace layout (~2.9 MB total, all 32B-aligned)
    uint8_t*  ws  = (uint8_t*)d_ws;
    int*      idx = (int*)ws;                                   //  393216 B
    float*    wt  = (float*)(ws + 393216);                      //  393216 B
    uint16_t* w1s = (uint16_t*)(ws + 786432);                   // 1572864 B
    uint16_t* w2s = (uint16_t*)(ws + 786432 + 1572864);         //  524288 B
    float*    bn  = (float*)(ws + 786432 + 1572864 + 524288);   //    8192 B

    k_bn<<<2, 256, 0, stream>>>(b1, g1, be1, m1, v1, b2, g2, be2, m2, v2, bn);
    k_swz<<<(512 * 1536 + 255) / 256, 256, 0, stream>>>(w1, w1s, 1536, 512 * 1536);
    k_swz<<<(512 * 512 + 255) / 256, 256, 0, stream>>>(w2, w2s, 512, 512 * 512);
    k_knn<<<B_ * (N_ / 256), 256, 0, stream>>>(xyz, centers, idx, wt);
    k_fused<<<B_ * (N_ / MT_), 256, 0, stream>>>(H4, H8, H12, idx, wt, w1s, w2s, bn,
                                                 (float*)d_out);
}
